// HandcraftedMultiplierV2_57380763075148
// MI455X (gfx1250) — matrix operand order repaired and checked by reading, not compiled, not run on hardware
//
#include <hip/hip_runtime.h>
#include <hip/hip_bf16.h>

// ---------------------------------------------------------------------------
// Output depends only on the 12 low ids per row -> build a 4096-entry LUT
// with WMMA (kernel 1), then a pure memory-bound decode pass (kernel 2).
// All WMMA operands are staged in zero-padded f16 LDS rows (stride 64 halves)
// so fragment builds are unconditional 16-B ds_load_b128 ops.
// ---------------------------------------------------------------------------

typedef __attribute__((ext_vector_type(16))) _Float16 v16h;
typedef __attribute__((ext_vector_type(8)))  _Float16 v8h;
typedef __attribute__((ext_vector_type(8)))  float    v8f;

#define WAVES_PER_BLOCK 4
#define LUT_ROWS 4096
#define KPAD 64  // padded K extent (halves) for every staged matrix row

// ---- WMMA fragment loaders (gfx1250 wave32 layouts, ISA 7.12.2) -----------
// A 16x32 f16: lane holds row m = lane%16, h = lane/16.
//   halves 0..7  <-> K = kb + 8h + {0..7}
//   halves 8..15 <-> K = kb + 16 + 8h + {0..7}
__device__ __forceinline__ v16h load_a(const _Float16* row, int kb, int h) {
  v8h lo = *(const v8h*)(row + kb + 8 * h);
  v8h hi = *(const v8h*)(row + kb + 16 + 8 * h);
  v16h a;
#pragma unroll
  for (int i = 0; i < 8; ++i) {
    a[i] = lo[i];
    a[8 + i] = hi[i];
  }
  return a;
}

// B 32x16 f16: lane covers column n = n0 + lane%16;
//   halves 0..7  <-> K = kb + 8h + {0..7}
//   halves 8..15 <-> K = kb + 16 + 8h + {0..7}
__device__ __forceinline__ v16h load_b(const _Float16* row, int kb, int h) {
  v8h lo = *(const v8h*)(row + kb + 8 * h);
  v8h hi = *(const v8h*)(row + kb + 16 + 8 * h);
  v16h b;
#pragma unroll
  for (int i = 0; i < 8; ++i) {
    b[i] = lo[i];
    b[8 + i] = hi[i];
  }
  return b;
}

__device__ __forceinline__ v8f wmma_f16(v16h a, v16h b, v8f c) {
  return __builtin_amdgcn_wmma_f32_16x16x32_f16(false, a, false, b, (short)0, c,
                                                false, false);
}

// ---------------------------------------------------------------------------
// Kernel 1: evaluate the collapsed network for all 4096 bit patterns.
// 64 blocks x 128 threads (4 waves); each wave = one 16-row tile.
// Per tile: 6 + 6 + 2 = 14 v_wmma_f32_16x16x32_f16.
// ---------------------------------------------------------------------------
__global__ __launch_bounds__(128) void lut_kernel(
    const float* __restrict__ emb, const float* __restrict__ Wv,
    const float* __restrict__ Wo, const float* __restrict__ W1,
    const float* __restrict__ b1g, const float* __restrict__ W2,
    const float* __restrict__ b2g, int* __restrict__ lut) {
  // f16 weight images, zero-padded to 48 x 64 (padding = the K/N clamps)
  __shared__ _Float16 sWoH[48 * KPAD];  // [n][K], n<36, K<36 valid
  __shared__ _Float16 sW1H[48 * KPAD];  // [f][K], K<36 valid
  __shared__ _Float16 sW2H[48 * KPAD];  // [e][K], e<36, K<48 valid
  // per-wave f16 activation rows, zero-padded to 64
  __shared__ _Float16 sC[WAVES_PER_BLOCK][16][KPAD];
  __shared__ _Float16 sAttnH[WAVES_PER_BLOCK][16][KPAD];
  __shared__ _Float16 sFH[WAVES_PER_BLOCK][16][KPAD];
  __shared__ float sP[2][36];  // P[t][e] = emb[t] . W_v[e]
  __shared__ float sb1[48];
  __shared__ float sb2[36];
  __shared__ float sTile[WAVES_PER_BLOCK][16][17];  // layer-3 f32 tile

  const int tid = threadIdx.x;

  // stage weights -> f16 LDS (padding written as zero, so no guards later)
  for (int i = tid; i < 48 * KPAD; i += 128) {
    const int n = i >> 6, K = i & (KPAD - 1);
    sWoH[i] = (_Float16)((n < 36 && K < 36) ? Wo[n * 36 + K] : 0.f);
    sW1H[i] = (_Float16)((K < 36) ? W1[n * 36 + K] : 0.f);
    sW2H[i] = (_Float16)((n < 36 && K < 48) ? W2[n * 48 + K] : 0.f);
  }
  // zero activation padding regions (full clear: simplest & branch-free)
  {
    unsigned* zA = (unsigned*)&sAttnH[0][0][0];
    unsigned* zF = (unsigned*)&sFH[0][0][0];
    for (int i = tid; i < WAVES_PER_BLOCK * 16 * KPAD / 2; i += 128) {
      zA[i] = 0u;
      zF[i] = 0u;
    }
  }
  if (tid < 48) sb1[tid] = b1g[tid];
  if (tid < 36) sb2[tid] = b2g[tid];
  if (tid < 72) {
    const int t = tid / 36, e = tid % 36;
    float acc = 0.f;
    for (int d = 0; d < 36; ++d) acc += emb[t * 36 + d] * Wv[e * 36 + d];
    sP[t][e] = acc;
  }
  __syncthreads();

  const int wave = tid >> 5;
  const int lane = tid & 31;
  const int m = lane & 15;
  const int h = lane >> 4;
  const int nl = lane & 15;  // B/C column within tile
  const int tile = blockIdx.x * WAVES_PER_BLOCK + wave;

  // build c rows for this wave's 16 patterns: c[e] = P[(r >> (e/3)) & 1][e]
  for (int idx = lane; idx < 16 * KPAD; idx += 32) {
    const int mm = idx >> 6, e = idx & (KPAD - 1);
    const int r = tile * 16 + mm;
    sC[wave][mm][e] =
        (_Float16)((e < 36) ? sP[(r >> (e / 3)) & 1][e] : 0.f);
  }
  __syncthreads();

  // ---- layer 1: attn = c @ Wo^T ----
  {
    const _Float16* arow = &sC[wave][m][0];
    v16h a0 = load_a(arow, 0, h);
    v16h a1 = load_a(arow, 32, h);
#pragma unroll
    for (int nt = 0; nt < 3; ++nt) {
      const int nn = nt * 16 + nl;
      const _Float16* brow = &sWoH[nn * KPAD];
      v8f acc = {};
      acc = wmma_f16(a0, load_b(brow, 0, h), acc);
      acc = wmma_f16(a1, load_b(brow, 32, h), acc);
      if (nn < 36) {
#pragma unroll
        for (int v = 0; v < 8; ++v)
          sAttnH[wave][v + 8 * h][nn] = (_Float16)acc[v];
      }
    }
  }
  __syncthreads();

  // ---- layer 2: f = relu(attn @ W1^T + b1) ----
  {
    const _Float16* arow = &sAttnH[wave][m][0];
    v16h a0 = load_a(arow, 0, h);
    v16h a1 = load_a(arow, 32, h);
#pragma unroll
    for (int nt = 0; nt < 3; ++nt) {
      const int nn = nt * 16 + nl;
      const _Float16* brow = &sW1H[nn * KPAD];
      v8f acc = {};
      acc = wmma_f16(a0, load_b(brow, 0, h), acc);
      acc = wmma_f16(a1, load_b(brow, 32, h), acc);
      const float bias = sb1[nn];
#pragma unroll
      for (int v = 0; v < 8; ++v)
        sFH[wave][v + 8 * h][nn] = (_Float16)fmaxf(acc[v] + bias, 0.f);
    }
  }
  __syncthreads();

  // ---- layer 3 (only h2[:,12:24] needed): mlp = f @ W2^T, n in [12,28) ----
  {
    const _Float16* arow = &sFH[wave][m][0];
    v16h a0 = load_a(arow, 0, h);
    v16h a1 = load_a(arow, 32, h);
    const int nn = 12 + nl;
    const _Float16* brow = &sW2H[nn * KPAD];
    v8f acc = {};
    acc = wmma_f16(a0, load_b(brow, 0, h), acc);
    acc = wmma_f16(a1, load_b(brow, 32, h), acc);
#pragma unroll
    for (int v = 0; v < 8; ++v) sTile[wave][v + 8 * h][nl] = acc[v];
  }
  __syncthreads();

  // ---- reduce: total = sum_j (attn+mlp+b2)[12+j] * 2^j, truncate to int ----
  if (lane < 16) {
    float tot = 0.f, p2 = 1.f;
#pragma unroll
    for (int j = 0; j < 12; ++j) {
      const float val = sTile[wave][lane][j] +
                        (float)sAttnH[wave][lane][12 + j] + sb2[12 + j];
      tot += val * p2;
      p2 *= 2.f;
    }
    lut[tile * 16 + lane] = (int)tot;
  }
}

// ---------------------------------------------------------------------------
// Kernel 2: memory-bound decode. Per row: 48 B of ids in, 192 B out.
// ~15.7 MB total -> ~0.7 us at 23.3 TB/s.
// ---------------------------------------------------------------------------
__global__ __launch_bounds__(256) void decode_kernel(
    const int* __restrict__ ids, const int* __restrict__ lut,
    float* __restrict__ out, int B) {
  const int b = blockIdx.x * 256 + threadIdx.x;
  if (b >= B) return;

  const int4* row = reinterpret_cast<const int4*>(ids) + (size_t)b * 6;
  const int4 w0 = row[0], w1 = row[1], w2 = row[2];
  const int r = (w0.x & 1)        | ((w0.y & 1) << 1)  | ((w0.z & 1) << 2) |
                ((w0.w & 1) << 3) | ((w1.x & 1) << 4)  | ((w1.y & 1) << 5) |
                ((w1.z & 1) << 6) | ((w1.w & 1) << 7)  | ((w2.x & 1) << 8) |
                ((w2.y & 1) << 9) | ((w2.z & 1) << 10) | ((w2.w & 1) << 11);
  const int t = lut[r];

  float4* o = reinterpret_cast<float4*>(out) + (size_t)b * 12;

  // l = 0..11: ki = 0 -> bit 0 of total
  const float bit0 = (float)(t & 1);
  const float x0 = 10.f * bit0 - 0.5f;
  const float4 p0 = {-x0, x0, -x0, x0};
#pragma unroll
  for (int q = 0; q < 6; ++q) o[q] = p0;

  // l = 12..22: ki = l-11; l = 23: masked to zero
#pragma unroll
  for (int q = 0; q < 6; ++q) {
    const int lA = 12 + 2 * q;  // 12,14,...,22
    const int lB = lA + 1;      // 13,15,...,23
    const float bA = (float)((t >> (lA - 11)) & 1);
    const float xA = 10.f * bA - 0.5f;
    float4 v;
    v.x = -xA;
    v.y = xA;
    if (lB < 23) {
      const float bB = (float)((t >> (lB - 11)) & 1);
      const float xB = 10.f * bB - 0.5f;
      v.z = -xB;
      v.w = xB;
    } else {
      v.z = 0.f;
      v.w = 0.f;
    }
    o[6 + q] = v;
  }
}

// ---------------------------------------------------------------------------
extern "C" void kernel_launch(void* const* d_in, const int* in_sizes, int n_in,
                              void* d_out, int out_size, void* d_ws,
                              size_t ws_size, hipStream_t stream) {
  const int*   ids = (const int*)d_in[0];
  const float* emb = (const float*)d_in[1];
  const float* Wv  = (const float*)d_in[2];
  const float* Wo  = (const float*)d_in[3];
  const float* W1  = (const float*)d_in[4];
  const float* b1  = (const float*)d_in[5];
  const float* W2  = (const float*)d_in[6];
  const float* b2  = (const float*)d_in[7];
  float* out = (float*)d_out;
  int*   lut = (int*)d_ws;  // 4096 * 4 B = 16 KB of scratch

  const int B = in_sizes[0] / 24;

  lut_kernel<<<LUT_ROWS / (WAVES_PER_BLOCK * 16), 128, 0, stream>>>(
      emb, Wv, Wo, W1, b1, W2, b2, lut);
  decode_kernel<<<(B + 255) / 256, 256, 0, stream>>>(ids, lut, out, B);
}
